// Residual_block_40484361732782
// MI455X (gfx1250) — matrix operand order repaired and checked by reading, not compiled, not run on hardware
//
#include <hip/hip_runtime.h>
#include <hip/hip_bf16.h>

// ---------------------------------------------------------------------------
// CDNA5 / gfx1250 wave32 WMMA implementation of the STGCN residual block.
//  B=32, N=1000, C=32, T=24, K=3, F_CHEB=F_TIME=64
//  All contractions run on v_wmma_f32_16x16x32_bf16.
//  gemm1 (87% of FLOPs): 128x128x32 block tile, double-buffered LDS,
//  async global->LDS staging (global_load_async_to_lds_b128 + ASYNCcnt).
// ---------------------------------------------------------------------------

typedef __attribute__((ext_vector_type(16))) __bf16 v16bf;
typedef __attribute__((ext_vector_type(8)))  float  v8f;
typedef __attribute__((ext_vector_type(4)))  int    v4i;

struct alignas(16) U128 { unsigned int a, b, c, d; };

union FragBF {
    v16bf v;
    U128  u[2];
};

__device__ __forceinline__ v8f wmma_bf16(v16bf a, v16bf b, v8f c) {
    // (neg_a, A, neg_b, B, c_mod, C, reuse_a, reuse_b)
    return __builtin_amdgcn_wmma_f32_16x16x32_bf16(false, a, false, b,
                                                   (short)0, c, false, false);
}

// --- async global->LDS staging (CDNA5), with safe fallback ----------------
#if defined(__gfx1250__) && \
    __has_builtin(__builtin_amdgcn_global_load_async_to_lds_b128) && \
    __has_builtin(__builtin_amdgcn_s_wait_asynccnt)
#define GCN_ASYNC 1
#else
#define GCN_ASYNC 0
#endif

__device__ __forceinline__ void cp16(__bf16* l, const __bf16* g) {
#if GCN_ASYNC
    __builtin_amdgcn_global_load_async_to_lds_b128(
        (__attribute__((address_space(1))) v4i*)(g),
        (__attribute__((address_space(3))) v4i*)(l), 0, 0);
#else
    *(U128*)l = *(const U128*)g;
#endif
}

__device__ __forceinline__ void wait_async0() {
#if GCN_ASYNC
    __builtin_amdgcn_s_wait_asynccnt(0);
#endif
}

// ---------------------------------------------------------------------------
// Workspace layout (bytes, 256-aligned):
//   Ab      @ 0           : 3*1024*1024   bf16  (cheb, K padded to 1024)
//   XbT     @ 6291456     : 24576*1024    bf16  (x as B-operand, [j][m])
//   Y2      @ 56623104    : 768000*96     bf16  (graph-conv result [p][(k,c)])
//   spz     @ 204079104   : 32*1000*26*64 bf16  (relu'd spatial, time-padded)
//   xTb     @ 310575104   : 768000*32     bf16  (x as residual A, [p][c])
//   ThetaTb @ 359727104   : 64*96         bf16  ([f][q])
//   W3b     @ 359739392   : 64*192        bf16  ([fo][dt*64+fi])
//   resWb   @ 359763968   : 64*32         bf16  ([fo][c])
// ---------------------------------------------------------------------------

__global__ void pack_cheb_kernel(const float* __restrict__ cheb,
                                 __bf16* __restrict__ Ab) {
    unsigned idx = blockIdx.x * 256u + threadIdx.x;          // 3*1024*1024
    unsigned k = idx >> 20, n = (idx >> 10) & 1023u, m = idx & 1023u;
    float v = 0.f;
    if (n < 1000u && m < 1000u) v = cheb[((size_t)k * 1000u + n) * 1000u + m];
    Ab[idx] = (__bf16)v;
}

__global__ void pack_xbt_kernel(const float* __restrict__ x,
                                __bf16* __restrict__ XbT) {
    unsigned idx = blockIdx.x * 256u + threadIdx.x;          // 24576*1024
    unsigned j = idx >> 10, m = idx & 1023u;
    float v = 0.f;
    if (m < 1000u) {
        unsigned b = j / 768u, r = j % 768u, c = r / 24u, t = r % 24u;
        v = x[(((size_t)b * 1000u + m) * 32u + c) * 24u + t];
    }
    XbT[idx] = (__bf16)v;
}

__global__ void pack_xt_kernel(const float* __restrict__ x,
                               __bf16* __restrict__ xTb) {
    unsigned idx = blockIdx.x * 256u + threadIdx.x;          // 768000*32
    unsigned p = idx >> 5, c = idx & 31u;
    unsigned b = p / 24000u, r = p % 24000u, n = r / 24u, t = r % 24u;
    xTb[idx] = (__bf16)x[(((size_t)b * 1000u + n) * 32u + c) * 24u + t];
}

__global__ void pack_w_kernel(const float* __restrict__ Theta,
                              const float* __restrict__ time_w,
                              const float* __restrict__ res_w,
                              __bf16* __restrict__ ThetaTb,
                              __bf16* __restrict__ W3b,
                              __bf16* __restrict__ resWb) {
    unsigned idx = blockIdx.x * 256u + threadIdx.x;
    if (idx < 6144u) {                                       // ThetaTb[f][q]
        unsigned f = idx / 96u, q = idx % 96u, k = q >> 5, c = q & 31u;
        ThetaTb[idx] = (__bf16)Theta[((k * 32u + c) * 64u) + f];
    } else if (idx < 6144u + 12288u) {                       // W3b[fo][dt*64+fi]
        unsigned i2 = idx - 6144u;
        unsigned fo = i2 / 192u, q = i2 % 192u, dt = q >> 6, fi = q & 63u;
        W3b[i2] = (__bf16)time_w[(fo * 64u + fi) * 3u + dt];
    } else if (idx < 6144u + 12288u + 2048u) {               // resWb[fo][c]
        unsigned i3 = idx - 6144u - 12288u;
        resWb[i3] = (__bf16)res_w[i3];
    }
}

__global__ void zero_pad_kernel(__bf16* __restrict__ spz) {
    unsigned i = blockIdx.x * 256u + threadIdx.x;            // 512000
    unsigned rowi = i >> 3, off = i & 7u;
    unsigned edge = rowi & 1u, bn = rowi >> 1;               // bn < 32000
    unsigned t1 = edge ? 25u : 0u;
    U128 z; z.a = z.b = z.c = z.d = 0u;
    *(U128*)(spz + ((size_t)bn * 26u + t1) * 64u + off * 8u) = z;
}

// ---------------------------------------------------------------------------
// gemm1: Y[k][n][j=(b,c,t)] = sum_m cheb[k][n][m] * x[m][j]
// BM=128, BN=128, BK=32, 256 threads (8 waves), 8 accumulators / wave.
// Double-buffered LDS; async global->LDS staging on CDNA5.
// Writes bf16 to Y2[p=(b,n,t)][q=k*32+c].
// ---------------------------------------------------------------------------
__global__ void __launch_bounds__(256) gemm1_kernel(const __bf16* __restrict__ Ab,
                                                    const __bf16* __restrict__ XbT,
                                                    __bf16* __restrict__ Y2) {
    // row stride 40 elems (80B) breaks LDS bank conflicts; 16B-aligned chunks.
    __shared__ alignas(16) __bf16 As[2][128 * 40];
    __shared__ alignas(16) __bf16 Bs[2][128 * 40];

    const unsigned tid  = threadIdx.x;
    const unsigned kz   = blockIdx.z;
    const unsigned n0   = blockIdx.y * 128u;
    const unsigned j0   = blockIdx.x * 128u;
    const unsigned wave = tid >> 5, lane = tid & 31u;
    const unsigned l16  = lane & 15u;
    const bool     hi   = lane >= 16u;

    v8f acc[8] = {};

    // staging: each thread moves 32B of A and 32B of B per K-step
    const unsigned row  = tid >> 1;                 // 0..127
    const unsigned seg  = (tid & 1u) * 16u;         // 0 or 16 (elems)
    const size_t   aBase = ((size_t)kz * 1024u + n0 + row) * 1024u + seg;
    const size_t   bBase = ((size_t)(j0 + row)) * 1024u + seg;
    const unsigned lOff  = row * 40u + seg;

    // prologue: stage tile 0 into buffer 0
    cp16(&As[0][lOff],      Ab  + aBase);
    cp16(&As[0][lOff + 8u], Ab  + aBase + 8u);
    cp16(&Bs[0][lOff],      XbT + bBase);
    cp16(&Bs[0][lOff + 8u], XbT + bBase + 8u);

    unsigned cur = 0u;
    for (unsigned i = 0; i < 32u; ++i) {
        wait_async0();            // my async stores into As[cur]/Bs[cur] done
        __syncthreads();          // everyone's done; old buffer free to reuse

        if (i + 1u < 32u) {       // stage next tile into the other buffer,
            size_t m0 = (size_t)(i + 1u) * 32u;      // overlapping the WMMAs
            unsigned nb = cur ^ 1u;
            cp16(&As[nb][lOff],      Ab  + aBase + m0);
            cp16(&As[nb][lOff + 8u], Ab  + aBase + m0 + 8u);
            cp16(&Bs[nb][lOff],      XbT + bBase + m0);
            cp16(&Bs[nb][lOff + 8u], XbT + bBase + m0 + 8u);
        }

        FragBF bf;
        {
            unsigned cb = (wave * 16u + l16) * 40u + (hi ? 16u : 0u);
            bf.u[0] = *(const U128*)&Bs[cur][cb];
            bf.u[1] = *(const U128*)&Bs[cur][cb + 8u];
        }
#pragma unroll
        for (unsigned r = 0; r < 8u; ++r) {
            FragBF af;
            unsigned rb = (r * 16u + l16) * 40u;
            af.u[0] = *(const U128*)&As[cur][rb + (hi ? 16u : 0u)];
            af.u[1] = *(const U128*)&As[cur][rb + 8u + (hi ? 16u : 0u)];
            acc[r] = wmma_bf16(af.v, bf.v, acc[r]);
        }
        cur ^= 1u;
    }

    // scatter-store bf16 into Y2[p][q]
    const unsigned j = j0 + wave * 16u + l16;
    const unsigned b = j / 768u, jr = j % 768u, c = jr / 24u, t = jr % 24u;
    const unsigned q = kz * 32u + c;
#pragma unroll
    for (unsigned r = 0; r < 8u; ++r) {
#pragma unroll
        for (unsigned e = 0; e < 8u; ++e) {
            unsigned n = n0 + r * 16u + e + (hi ? 8u : 0u);
            if (n < 1000u) {
                unsigned p = (b * 1000u + n) * 24u + t;
                Y2[(size_t)p * 96u + q] = (__bf16)acc[r][e];
            }
        }
    }
}

// ---------------------------------------------------------------------------
// gemm2: spatial[p][f] = relu( sum_q Y2[p][q] * Theta[q][f] ),  Kdim=96.
// One wave per 16-row p-tile, 4 f-tiles, 3 WMMA k-steps. Writes time-padded
// bf16 spz[b][n][t+1][f].
// ---------------------------------------------------------------------------
__global__ void __launch_bounds__(256) gemm2_kernel(const __bf16* __restrict__ Y2,
                                                    const __bf16* __restrict__ ThetaTb,
                                                    __bf16* __restrict__ spz) {
    const unsigned tid = threadIdx.x, wave = tid >> 5, lane = tid & 31u;
    const unsigned l16 = lane & 15u;
    const bool     hi  = lane >= 16u;
    const unsigned p0  = (blockIdx.x * 8u + wave) * 16u;
    const __bf16*  Yp  = Y2 + (size_t)(p0 + l16) * 96u;

    v8f acc[4] = {};
#pragma unroll
    for (unsigned s = 0; s < 3u; ++s) {
        FragBF af;
        af.u[0] = *(const U128*)(Yp + s * 32u + (hi ? 16u : 0u));
        af.u[1] = *(const U128*)(Yp + s * 32u + 8u + (hi ? 16u : 0u));
#pragma unroll
        for (unsigned ft = 0; ft < 4u; ++ft) {
            unsigned f = ft * 16u + l16;
            FragBF bf;
            const __bf16* Tp = ThetaTb + f * 96u + s * 32u + (hi ? 16u : 0u);
            bf.u[0] = *(const U128*)Tp;
            bf.u[1] = *(const U128*)(Tp + 8u);
            acc[ft] = wmma_bf16(af.v, bf.v, acc[ft]);
        }
    }
#pragma unroll
    for (unsigned ft = 0; ft < 4u; ++ft) {
        unsigned f = ft * 16u + l16;
#pragma unroll
        for (unsigned e = 0; e < 8u; ++e) {
            unsigned p = p0 + e + (hi ? 8u : 0u);
            unsigned b = p / 24000u, pr = p % 24000u, n = pr / 24u, t = pr % 24u;
            float v = acc[ft][e];
            v = v > 0.f ? v : 0.f;
            spz[(((size_t)b * 1000u + n) * 26u + t + 1u) * 64u + f] = (__bf16)v;
        }
    }
}

// ---------------------------------------------------------------------------
// gemm3: time conv (Kdim=192 over (dt,fi), 6 WMMA) + 1x1 residual (Kdim=32,
// 1 WMMA) + bias + tanh*sigmoid gating, f32 output in [B,N,F,T].
// ---------------------------------------------------------------------------
__global__ void __launch_bounds__(256) gemm3_kernel(const __bf16* __restrict__ spz,
                                                    const __bf16* __restrict__ xTb,
                                                    const __bf16* __restrict__ W3b,
                                                    const __bf16* __restrict__ resWb,
                                                    const float* __restrict__ time_b,
                                                    const float* __restrict__ res_b,
                                                    float* __restrict__ out) {
    const unsigned tid = threadIdx.x, wave = tid >> 5, lane = tid & 31u;
    const unsigned l16 = lane & 15u;
    const bool     hi  = lane >= 16u;
    const unsigned p0  = (blockIdx.x * 8u + wave) * 16u;

    const unsigned pl  = p0 + l16;
    const unsigned bl  = pl / 24000u, plr = pl % 24000u;
    const unsigned nl  = plr / 24u,   tl  = plr % 24u;
    const size_t   rowbase = ((size_t)bl * 1000u + nl) * 26u;

    v8f tc[4] = {}, rs[4] = {};

#pragma unroll
    for (unsigned s = 0; s < 6u; ++s) {
        unsigned dt = s >> 1, fib = (s & 1u) * 32u;
        const __bf16* Ar = spz + (rowbase + tl + dt) * 64u + fib;
        FragBF af;
        af.u[0] = *(const U128*)(Ar + (hi ? 16u : 0u));
        af.u[1] = *(const U128*)(Ar + 8u + (hi ? 16u : 0u));
#pragma unroll
        for (unsigned ft = 0; ft < 4u; ++ft) {
            unsigned f = ft * 16u + l16;
            FragBF bf;
            const __bf16* Wp = W3b + f * 192u + s * 32u + (hi ? 16u : 0u);
            bf.u[0] = *(const U128*)Wp;
            bf.u[1] = *(const U128*)(Wp + 8u);
            tc[ft] = wmma_bf16(af.v, bf.v, tc[ft]);
        }
    }
    {   // residual 1x1, Kdim = 32
        FragBF af;
        const __bf16* Xr = xTb + (size_t)pl * 32u;
        af.u[0] = *(const U128*)(Xr + (hi ? 16u : 0u));
        af.u[1] = *(const U128*)(Xr + 8u + (hi ? 16u : 0u));
#pragma unroll
        for (unsigned ft = 0; ft < 4u; ++ft) {
            unsigned f = ft * 16u + l16;
            FragBF bf;
            const __bf16* Rp = resWb + f * 32u + (hi ? 16u : 0u);
            bf.u[0] = *(const U128*)Rp;
            bf.u[1] = *(const U128*)(Rp + 8u);
            rs[ft] = wmma_bf16(af.v, bf.v, rs[ft]);
        }
    }
#pragma unroll
    for (unsigned ft = 0; ft < 4u; ++ft) {
        unsigned f = ft * 16u + l16;
        float tb = time_b[f], rb = res_b[f];
#pragma unroll
        for (unsigned e = 0; e < 8u; ++e) {
            unsigned p = p0 + e + (hi ? 8u : 0u);
            unsigned b = p / 24000u, pr = p % 24000u, n = pr / 24u, t = pr % 24u;
            float tcv = tc[ft][e] + tb;
            float sg  = 1.0f / (1.0f + __expf(-tcv));
            float g   = tanhf(tcv) * sg;
            out[(((size_t)b * 1000u + n) * 64u + f) * 24u + t] = rs[ft][e] + rb + g;
        }
    }
}

// ---------------------------------------------------------------------------
extern "C" void kernel_launch(void* const* d_in, const int* in_sizes, int n_in,
                              void* d_out, int out_size, void* d_ws, size_t ws_size,
                              hipStream_t stream) {
    const float* x      = (const float*)d_in[0];
    const float* cheb   = (const float*)d_in[1];
    const float* Theta  = (const float*)d_in[2];
    const float* time_w = (const float*)d_in[3];
    const float* time_b = (const float*)d_in[4];
    const float* res_w  = (const float*)d_in[5];
    const float* res_b  = (const float*)d_in[6];
    float* out = (float*)d_out;

    char* ws = (char*)d_ws;
    __bf16* Ab      = (__bf16*)(ws + 0ull);
    __bf16* XbT     = (__bf16*)(ws + 6291456ull);
    __bf16* Y2      = (__bf16*)(ws + 56623104ull);
    __bf16* spz     = (__bf16*)(ws + 204079104ull);
    __bf16* xTb     = (__bf16*)(ws + 310575104ull);
    __bf16* ThetaTb = (__bf16*)(ws + 359727104ull);
    __bf16* W3b     = (__bf16*)(ws + 359739392ull);
    __bf16* resWb   = (__bf16*)(ws + 359763968ull);

    pack_cheb_kernel<<<12288, 256, 0, stream>>>(cheb, Ab);
    pack_xbt_kernel<<<98304, 256, 0, stream>>>(x, XbT);
    pack_xt_kernel<<<96000, 256, 0, stream>>>(x, xTb);
    pack_w_kernel<<<80, 256, 0, stream>>>(Theta, time_w, res_w, ThetaTb, W3b, resWb);
    zero_pad_kernel<<<2000, 256, 0, stream>>>(spz);

    gemm1_kernel<<<dim3(192, 8, 3), 256, 0, stream>>>(Ab, XbT, Y2);
    gemm2_kernel<<<6000, 256, 0, stream>>>(Y2, ThetaTb, spz);
    gemm3_kernel<<<6000, 256, 0, stream>>>(spz, xTb, W3b, resWb, time_b, res_b, out);
}
